// MultiHeadedAttention_31980326486610
// MI455X (gfx1250) — matrix operand order repaired and checked by reading, not compiled, not run on hardware
//
#include <hip/hip_runtime.h>
#include <hip/hip_bf16.h>

// ---------------------------------------------------------------------------
// MultiHeadedAttention (channel-wise outer-product attention) for MI455X
// N=4096 rows, D=512, H=8 heads, d_k=64.
//   q = x_q@Wq+bq ; k = x_k@Wk+bk ; v = x_v@Wv+bv        (bf16 WMMA GEMMs)
//   x[n,h,i] = sum_j softmax_j(q_i*k_j/8) * v_j           (VALU softmax)
//   out = (x_q - x)@Wo + bo                               (bf16 WMMA GEMM)
// A-tile staging uses the CDNA5 Tensor Data Mover (tensor_load_to_lds).
// ---------------------------------------------------------------------------

typedef __attribute__((ext_vector_type(16))) __bf16 v16bf;
typedef __attribute__((ext_vector_type(8)))  float  v8f;
typedef unsigned int v4u __attribute__((ext_vector_type(4)));
typedef int          v8i __attribute__((ext_vector_type(8)));
typedef int          v4i __attribute__((ext_vector_type(4)));

#if defined(__has_builtin)
#if __has_builtin(__builtin_amdgcn_tensor_load_to_lds) && \
    __has_builtin(__builtin_amdgcn_s_wait_tensorcnt)
#define USE_TDM 1
#endif
#endif
#ifndef USE_TDM
#define USE_TDM 0
#endif

union FragU {
    struct { uint4 lo, hi; } q;   // 32 bytes
    v16bf bf;
};

__device__ __forceinline__ unsigned short f32_to_bf16_rne(float f) {
    unsigned int u = __float_as_uint(f);
    unsigned int rounded = u + 0x7FFFu + ((u >> 16) & 1u);
    return (unsigned short)(rounded >> 16);
}

#if USE_TDM
// Issue a TDM load of a 2D bf16 tile (tile_d0 x tile_d1, dim0 contiguous)
// from a (tensor_d0 x tensor_d1) row-major tensor into LDS at lds_byte_off.
// D# built per CDNA5 ISA 8.3-8.6: group0 {count,lds,global,type=2},
// group1 {data_size=2B, dims, tile dims, dim0 stride}, groups 2/3 unused.
// This toolchain's builtin takes 6 args (g0 v4u, g1 v8i, g2 v4i, g3 v4i,
// extra v8i, cpol); the 5th group is unused for a 2D tile (VADDR4 = NULL).
__device__ __forceinline__ void tdm_load_2d_bf16(
    unsigned lds_byte_off, const unsigned short* gptr,
    unsigned tensor_d0, unsigned tensor_d1,
    unsigned tile_d0, unsigned tile_d1, unsigned stride0_elems) {
    unsigned long long ga = (unsigned long long)gptr;
    v4u g0;
    g0[0] = 1u;                                   // count=1, user descriptor
    g0[1] = lds_byte_off;                         // lds_addr (bytes)
    g0[2] = (unsigned)(ga & 0xFFFFFFFFu);         // global_addr[31:0]
    g0[3] = (unsigned)((ga >> 32) & 0x1FFFFFFu)   // global_addr[56:32]
            | (2u << 30);                         // type = 2 ("image")
    v8i g1;
    g1[0] = (int)(1u << 16);                      // wg_mask=0, data_size=1(2B)
    g1[1] = (int)((tensor_d0 & 0xFFFFu) << 16);   // tensor_dim0[15:0]
    g1[2] = (int)((tensor_d0 >> 16) | ((tensor_d1 & 0xFFFFu) << 16));
    g1[3] = (int)((tensor_d1 >> 16) & 0xFFFFu) | (int)((tile_d0 & 0xFFFFu) << 16);
    g1[4] = (int)(tile_d1 & 0xFFFFu);             // tile_dim1; tile_dim2=0
    g1[5] = (int)stride0_elems;                   // tensor_dim0_stride[31:0]
    g1[6] = 0;                                    // stride0 hi / stride1 lo
    g1[7] = 0;                                    // stride1 hi
    v4i g2 = {0, 0, 0, 0};                        // dims 2/3 unused
    v4i g3 = {0, 0, 0, 0};
    v8i g4 = {0, 0, 0, 0, 0, 0, 0, 0};            // unused trailing group
    __builtin_amdgcn_tensor_load_to_lds(g0, g1, g2, g3, g4, 0);
}
#endif

// ------------------------- f32 -> bf16 convert -----------------------------
__global__ void cvt_f32_bf16_kernel(const float* __restrict__ in,
                                    unsigned short* __restrict__ out, int n) {
    for (int i = blockIdx.x * blockDim.x + threadIdx.x; i < n;
         i += gridDim.x * blockDim.x) {
        out[i] = f32_to_bf16_rne(in[i]);
    }
}

// ------------------------- bf16 WMMA GEMM ----------------------------------
// C[M,N] = A[M,K] (bf16, row-major) x B[K,N] (bf16, row-major) + bias[N]
// Block: 256 threads = 8 wave32s as 4(M) x 2(N). Block tile 128x64.
// Each wave computes a 32x32 patch = 2x2 tiles of v_wmma_f32_16x16x32_bf16.
// A tile (128x32) is DMA'd into LDS by the Tensor Data Mover (wave 0),
// B tile (32x64) is staged transposed so fragment reads are b128 loads.
__global__ __launch_bounds__(256)
void gemm_bf16_wmma_kernel(const unsigned short* __restrict__ A,
                           const unsigned short* __restrict__ B,
                           const float* __restrict__ bias,
                           float* __restrict__ C,
                           int M, int N, int K) {
    __shared__ __align__(16) unsigned short Alds[128][32];
    __shared__ __align__(16) unsigned short Btlds[64][32];

    const int tid   = threadIdx.x;
    const int lane  = tid & 31;
    const int wave  = tid >> 5;
    const int wm    = wave >> 1;   // 0..3 : 32-row strip within block
    const int wn    = wave & 1;    // 0..1 : 32-col strip within block
    const int m0    = blockIdx.y * 128;
    const int n0    = blockIdx.x * 64;
    const int khalf = lane >> 4;
    const int lrow  = lane & 15;

    v8f acc[2][2] = {};

    for (int k0 = 0; k0 < K; k0 += 32) {
        __syncthreads();   // previous iteration's LDS reads complete

#if USE_TDM
        // ---- stage A tile via Tensor Data Mover: 128x32 bf16 -> Alds
        if (wave == 0) {
            const unsigned aoff =
                (unsigned)(unsigned long long)(const void*)&Alds[0][0];
            tdm_load_2d_bf16(aoff, A + (size_t)m0 * K + k0,
                             (unsigned)K, (unsigned)M,
                             32u, 128u, (unsigned)K);
        }
#else
        // ---- fallback: manual A staging, 32 B per thread
        {
            const int row  = tid >> 1;
            const int half = tid & 1;
            const uint4* src = reinterpret_cast<const uint4*>(
                A + (size_t)(m0 + row) * K + k0 + half * 16);
            uint4* dst = reinterpret_cast<uint4*>(&Alds[row][half * 16]);
            dst[0] = src[0];
            dst[1] = src[1];
        }
#endif
        // ---- stage B tile transposed: 32 K-rows x 64 cols -> Bt[64][32]
        {
            const int krow  = tid >> 3;   // 0..31
            const int chunk = tid & 7;    // 8 cols per thread
            uint4 d = *reinterpret_cast<const uint4*>(
                B + (size_t)(k0 + krow) * N + n0 + chunk * 8);
            const unsigned short* ds16 =
                reinterpret_cast<const unsigned short*>(&d);
#pragma unroll
            for (int e = 0; e < 8; ++e)
                Btlds[chunk * 8 + e][krow] = ds16[e];
        }
#if USE_TDM
        if (wave == 0) __builtin_amdgcn_s_wait_tensorcnt(0);
#endif
        __syncthreads();

        // prefetch next A tile into cache (global_prefetch_b8)
        if (k0 + 32 < K) {
            __builtin_prefetch(A + (size_t)(m0 + (tid >> 1)) * K + (k0 + 32),
                               0, 1);
        }

        // ---- fragments + 4 WMMAs
        FragU a0, a1, b0, b1;
        {
            const int r0 = wm * 32 + lrow;
            a0.q.lo = *reinterpret_cast<const uint4*>(&Alds[r0][khalf * 8]);
            a0.q.hi = *reinterpret_cast<const uint4*>(&Alds[r0][16 + khalf * 8]);
            const int r1 = r0 + 16;
            a1.q.lo = *reinterpret_cast<const uint4*>(&Alds[r1][khalf * 8]);
            a1.q.hi = *reinterpret_cast<const uint4*>(&Alds[r1][16 + khalf * 8]);
            const int c0 = wn * 32 + lrow;
            b0.q.lo = *reinterpret_cast<const uint4*>(&Btlds[c0][khalf * 8]);
            b0.q.hi = *reinterpret_cast<const uint4*>(&Btlds[c0][16 + khalf * 8]);
            const int c1 = c0 + 16;
            b1.q.lo = *reinterpret_cast<const uint4*>(&Btlds[c1][khalf * 8]);
            b1.q.hi = *reinterpret_cast<const uint4*>(&Btlds[c1][16 + khalf * 8]);
        }
        acc[0][0] = __builtin_amdgcn_wmma_f32_16x16x32_bf16(
            false, a0.bf, false, b0.bf, (short)0, acc[0][0], false, false);
        acc[0][1] = __builtin_amdgcn_wmma_f32_16x16x32_bf16(
            false, a0.bf, false, b1.bf, (short)0, acc[0][1], false, false);
        acc[1][0] = __builtin_amdgcn_wmma_f32_16x16x32_bf16(
            false, a1.bf, false, b0.bf, (short)0, acc[1][0], false, false);
        acc[1][1] = __builtin_amdgcn_wmma_f32_16x16x32_bf16(
            false, a1.bf, false, b1.bf, (short)0, acc[1][1], false, false);
    }

    // ---- epilogue: bias add, fp32 store
#pragma unroll
    for (int ti = 0; ti < 2; ++ti) {
#pragma unroll
        for (int tj = 0; tj < 2; ++tj) {
            const int col = n0 + wn * 32 + tj * 16 + (lane & 15);
            const float bcol = bias[col];
            const int rowb = m0 + wm * 32 + ti * 16 + khalf * 8;
#pragma unroll
            for (int r = 0; r < 8; ++r) {
                C[(size_t)(rowb + r) * N + col] = acc[ti][tj][r] + bcol;
            }
        }
    }
}

// ------------------------- attention + residual ----------------------------
// One 64-thread group per (n,h) pair; 4 groups per 256-thread block.
// scores[i,j] = q_i * k_j / 8 ; p = softmax_j ; x_i = sum_j p[i,j] v_j
// r = x_q - x, emitted as bf16 for the final GEMM.
__global__ __launch_bounds__(256)
void attn_kernel(const float* __restrict__ Q, const float* __restrict__ K,
                 const float* __restrict__ V, const float* __restrict__ Xq,
                 unsigned short* __restrict__ Rbf) {
    __shared__ float kk[4][64];
    __shared__ float vv[4][64];

    const int g    = threadIdx.x >> 6;     // group 0..3
    const int i    = threadIdx.x & 63;     // channel within head
    const int pair = blockIdx.x * 4 + g;   // 0 .. 4096*8-1
    const int n    = pair >> 3;
    const int h    = pair & 7;
    const int base = n * 512 + h * 64;
    const float scale = 0.125f;            // 1/sqrt(64)

    const float q = Q[base + i];
    kk[g][i] = K[base + i];
    vv[g][i] = V[base + i];
    __syncthreads();

    float m = -3.402823466e+38f;
#pragma unroll 8
    for (int j = 0; j < 64; ++j)
        m = fmaxf(m, q * kk[g][j] * scale);

    float num = 0.f, den = 0.f;
#pragma unroll 8
    for (int j = 0; j < 64; ++j) {
        float e = __expf(q * kk[g][j] * scale - m);
        num += e * vv[g][j];
        den += e;
    }
    const float x = num / den;
    const float r = Xq[base + i] - x;
    Rbf[base + i] = f32_to_bf16_rne(r);
}

// ---------------------------------------------------------------------------
extern "C" void kernel_launch(void* const* d_in, const int* in_sizes, int n_in,
                              void* d_out, int out_size, void* d_ws,
                              size_t ws_size, hipStream_t stream) {
    (void)in_sizes; (void)n_in; (void)out_size; (void)ws_size;
    constexpr int NR = 4096, D = 512;

    const float* x_q = (const float*)d_in[0];
    const float* x_k = (const float*)d_in[1];
    const float* x_v = (const float*)d_in[2];
    const float* Wq  = (const float*)d_in[3];
    const float* bq  = (const float*)d_in[4];
    const float* Wk  = (const float*)d_in[5];
    const float* bk  = (const float*)d_in[6];
    const float* Wv  = (const float*)d_in[7];
    const float* bv  = (const float*)d_in[8];
    const float* Wo  = (const float*)d_in[9];
    const float* bo  = (const float*)d_in[10];
    float* out = (float*)d_out;

    // workspace layout (bytes)
    char* ws = (char*)d_ws;
    size_t off = 0;
    const size_t xN = (size_t)NR * D;   // 2,097,152
    const size_t wN = (size_t)D * D;    // 262,144
    unsigned short* Xq_bf = (unsigned short*)(ws + off); off += xN * 2;
    unsigned short* Xk_bf = (unsigned short*)(ws + off); off += xN * 2;
    unsigned short* Xv_bf = (unsigned short*)(ws + off); off += xN * 2;
    unsigned short* Wq_bf = (unsigned short*)(ws + off); off += wN * 2;
    unsigned short* Wk_bf = (unsigned short*)(ws + off); off += wN * 2;
    unsigned short* Wv_bf = (unsigned short*)(ws + off); off += wN * 2;
    unsigned short* Wo_bf = (unsigned short*)(ws + off); off += wN * 2;
    float* Qf = (float*)(ws + off); off += xN * 4;
    float* Kf = (float*)(ws + off); off += xN * 4;
    float* Vf = (float*)(ws + off); off += xN * 4;
    unsigned short* Rbf = (unsigned short*)(ws + off); off += xN * 2;

    // 1) convert activations + weights to bf16
    {
        const int blk = 256;
        cvt_f32_bf16_kernel<<<1024, blk, 0, stream>>>(x_q, Xq_bf, (int)xN);
        cvt_f32_bf16_kernel<<<1024, blk, 0, stream>>>(x_k, Xk_bf, (int)xN);
        cvt_f32_bf16_kernel<<<1024, blk, 0, stream>>>(x_v, Xv_bf, (int)xN);
        cvt_f32_bf16_kernel<<<512,  blk, 0, stream>>>(Wq,  Wq_bf, (int)wN);
        cvt_f32_bf16_kernel<<<512,  blk, 0, stream>>>(Wk,  Wk_bf, (int)wN);
        cvt_f32_bf16_kernel<<<512,  blk, 0, stream>>>(Wv,  Wv_bf, (int)wN);
        cvt_f32_bf16_kernel<<<512,  blk, 0, stream>>>(Wo,  Wo_bf, (int)wN);
    }

    // 2) QKV projections: C = X @ W + b  (M=4096, N=512, K=512)
    dim3 ggrid(D / 64, NR / 128, 1);   // (8, 32)
    gemm_bf16_wmma_kernel<<<ggrid, 256, 0, stream>>>(Xq_bf, Wq_bf, bq, Qf,
                                                     NR, D, D);
    gemm_bf16_wmma_kernel<<<ggrid, 256, 0, stream>>>(Xk_bf, Wk_bf, bk, Kf,
                                                     NR, D, D);
    gemm_bf16_wmma_kernel<<<ggrid, 256, 0, stream>>>(Xv_bf, Wv_bf, bv, Vf,
                                                     NR, D, D);

    // 3) channel-wise softmax attention + residual (bf16 out)
    attn_kernel<<<(NR * 8) / 4, 256, 0, stream>>>(Qf, Kf, Vf, x_q, Rbf);

    // 4) output projection: out = R @ Wo + bo
    gemm_bf16_wmma_kernel<<<ggrid, 256, 0, stream>>>(Rbf, Wo_bf, bo, out,
                                                     NR, D, D);
}
